// PhysicsSmootherSMPLX_10359461118494
// MI455X (gfx1250) — compile-verified, matrix operand order checked by reading, not run on hardware
//
#include <hip/hip_runtime.h>
#include <stdint.h>

// ---------------------------------------------------------------------------
// PhysicsSmootherSMPLX for MI455X (gfx1250, wave32)
//
// Memory-bound first-order IIR along T. Blocked (chunked) linear scan:
//   Phase A : per-chunk summaries  (P = prod tau, S = local scan w/ zero init)
//   Phase B : sequential combine across 16 chunks per channel (tiny)
//   Phase C : final per-chunk scan with correct incoming state -> output
// tau is computed once at Tp resolution; the Z=128 reduction uses WMMA
// (A x ones row-sum), tau chunk staging into LDS uses the Tensor Data Mover.
// ---------------------------------------------------------------------------

typedef __attribute__((ext_vector_type(16))) _Float16      v16h;
typedef __attribute__((ext_vector_type(8)))  float         v8f;
typedef __attribute__((ext_vector_type(4)))  unsigned int  v4u;
typedef __attribute__((ext_vector_type(8)))  int           v8i;
typedef __attribute__((ext_vector_type(4)))  int           v4i;

#define B_   32
#define T_   4096
#define J_   55
#define D_   6
#define JD_  330          // J_*D_
#define TP_  1024
#define Z_   128
#define NC_  16           // chunks per sequence
#define CT_  256          // timesteps per chunk
#define CTP_ 64           // tau rows per chunk (CT_/ratio, ratio=4)
#define TAUW (CTP_*J_)    // 3520 floats staged per chunk

// workspace layout (floats)
#define OFF_SIGMA 0
#define OFF_TAU   (B_*TP_)                       // 32768
#define OFF_P     (OFF_TAU + B_*TP_*J_)          // + 1,802,240
#define OFF_S     (OFF_P  + B_*J_*NC_)           // + 28,160
#define OFF_H     (OFF_S  + B_*J_*D_*NC_)        // + 168,960

__constant__ float c_mass[J_] = {
    0.1117f, 0.1416f, 0.1416f, 0.1633f, 0.0433f, 0.0433f, 0.1596f, 0.0137f,
    0.0137f, 0.1596f, 0.0137f, 0.0137f, 0.0350f, 0.0080f, 0.0080f, 0.0694f,
    0.0271f, 0.0271f, 0.0162f, 0.0162f, 0.0061f, 0.0061f, 0.0070f, 0.0010f,
    0.0010f,
    0.0004f, 0.0004f, 0.0004f, 0.0004f, 0.0004f, 0.0004f, 0.0004f, 0.0004f,
    0.0004f, 0.0004f, 0.0004f, 0.0004f, 0.0004f, 0.0004f, 0.0004f, 0.0004f,
    0.0004f, 0.0004f, 0.0004f, 0.0004f, 0.0004f, 0.0004f, 0.0004f, 0.0004f,
    0.0004f, 0.0004f, 0.0004f, 0.0004f, 0.0004f, 0.0004f
};

#if defined(__has_builtin)
#if __has_builtin(__builtin_amdgcn_tensor_load_to_lds) && __has_builtin(__builtin_amdgcn_s_wait_tensorcnt)
#define USE_TDM 1
#endif
#endif
#ifndef USE_TDM
#define USE_TDM 0
#endif

// ---------------------------------------------------------------------------
// Kernel 1: sigma[b,tp] = mean(exp(logvar[b,tp,:])) via WMMA row-sum.
// One wave handles 16 rows x 128 cols: 4 chained v_wmma_f32_16x16x32_f16 with
// an all-ones B matrix. Row sums are replicated across all N columns, so per
// the 16x16 f32 C/D layout lane 0 holds rows 0..7 and lane 16 rows 8..15.
// ---------------------------------------------------------------------------
__global__ __launch_bounds__(256) void k_sigma(const float* __restrict__ logvar,
                                               float* __restrict__ sigma) {
    const int lane    = threadIdx.x & 31;
    const int wave    = blockIdx.x * (blockDim.x >> 5) + (threadIdx.x >> 5);
    const int rowBase = wave * 16;                 // 2048 waves * 16 = 32768 rows
    const int m       = lane & 15;
    const int hi      = lane >> 4;
    const float* rp = logvar + (size_t)(rowBase + m) * Z_;

    v16h ones;
#pragma unroll
    for (int i = 0; i < 16; ++i) ones[i] = (_Float16)1.0f;

    v8f acc = {};
#pragma unroll
    for (int s = 0; s < 4; ++s) {
        // Per the 16-bit A 16x32 layout, lanes 0-15 carry K={0..7,16..23},
        // lanes 16-31 carry K={8..15,24..31}. (Row-sum is K-permutation
        // invariant; we only need disjoint coverage of the 32 columns.)
        const int c0 = s * 32 + (hi ? 8 : 0);
        const int c1 = s * 32 + 16 + (hi ? 8 : 0);
        v16h a;
#pragma unroll
        for (int k = 0; k < 8; ++k) a[k]     = (_Float16)__expf(rp[c0 + k]);
#pragma unroll
        for (int k = 0; k < 8; ++k) a[8 + k] = (_Float16)__expf(rp[c1 + k]);
        acc = __builtin_amdgcn_wmma_f32_16x16x32_f16(
            /*neg_a=*/false, a, /*neg_b=*/false, ones,
            /*c_mod=*/(short)0, acc, /*reuse_a=*/false, /*reuse_b=*/false);
    }
    if (m == 0) {  // lanes 0 and 16
        const float inv = 1.0f / (float)Z_;
#pragma unroll
        for (int r = 0; r < 8; ++r)
            sigma[rowBase + hi * 8 + r] = acc[r] * inv;
    }
}

// ---------------------------------------------------------------------------
// Kernel 2: tau[b,tp,j] = clip(0.5*shape[j]*(1-psi)*(1+sigma), 0, 0.95)
// ---------------------------------------------------------------------------
__global__ __launch_bounds__(256) void k_tau(const float* __restrict__ gate,
                                             const float* __restrict__ sigma,
                                             float* __restrict__ tau) {
    const int idx = blockIdx.x * blockDim.x + threadIdx.x;
    if (idx >= B_ * TP_ * J_) return;
    const int j  = idx % J_;
    const int bt = idx / J_;                         // b*TP_ + tp
    const float shape = fmaxf(sqrtf(c_mass[j] * (1.0f / 0.1633f)), 0.1f);
    float t = 0.5f * shape * (1.0f - gate[bt]) * (1.0f + sigma[bt]);
    tau[idx] = fminf(fmaxf(t, 0.0f), 0.95f);
}

// ---------------------------------------------------------------------------
// Stage one chunk's tau slice (contiguous 3520 floats) into LDS.
// Wave 0 issues a 1D TDM tensor_load_to_lds and waits TENSORcnt; barrier
// releases the other waves. Fallback: cooperative copy.
// ---------------------------------------------------------------------------
__device__ __forceinline__ void stage_tau(const float* __restrict__ tau_g,
                                          int b, int c, float* lds) {
    const size_t base = ((size_t)b * TP_ + (size_t)c * CTP_) * (size_t)J_;
#if USE_TDM
    if (threadIdx.x < 32) {                      // wave 0 only (wave-uniform)
        const uint64_t ga = (uint64_t)(uintptr_t)(tau_g + base);
        const uint32_t la = (uint32_t)(uintptr_t)lds;   // low 32 bits = LDS offset
        v4u g0;
        g0[0] = 1u;                               // count=1, user desc
        g0[1] = la;                               // lds_addr
        g0[2] = (uint32_t)ga;                     // global_addr[31:0]
        g0[3] = (uint32_t)(ga >> 32) | (2u << 30);// global_addr[56:32] | type=2
        v8i g1;
        g1[0] = (int)(2u << 16);                  // data_size = 4B code 2
        g1[1] = (int)((TAUW & 0xFFFFu) << 16);    // tensor_dim0[15:0]
        g1[2] = (int)((1u << 16) | (TAUW >> 16)); // tensor_dim0[31:16], tensor_dim1=1
        g1[3] = (int)((unsigned)TAUW << 16);      // tile_dim0 = 3520 (1D tile)
        g1[4] = 0;                                // tile_dim1 = 0 (unused), tile_dim2 = 0
        g1[5] = (int)TAUW;                        // tensor_dim0_stride[31:0]
        g1[6] = 0;                                // stride hi, tensor_dim1_stride = 0
        g1[7] = 0;
        const v4i z4 = {0, 0, 0, 0};
        const v8i z8 = {0, 0, 0, 0, 0, 0, 0, 0};
        __builtin_amdgcn_tensor_load_to_lds(g0, g1, z4, z4, z8, 0);
        __builtin_amdgcn_s_wait_tensorcnt((short)0);
    }
#else
    for (int i = threadIdx.x; i < TAUW; i += blockDim.x) lds[i] = tau_g[base + i];
#endif
    __syncthreads();
}

// ---------------------------------------------------------------------------
// Kernel 3 (Phase A): per-chunk summaries. Block = (b, chunk); thread = jd.
// Local scan with zero incoming state gives S; P = product of tau over chunk.
// ---------------------------------------------------------------------------
__global__ __launch_bounds__(352) void k_chunkA(const float* __restrict__ x,
                                                const float* __restrict__ tau,
                                                float* __restrict__ P,
                                                float* __restrict__ S) {
    __shared__ float lt[TAUW];
    const int b = blockIdx.x >> 4;
    const int c = blockIdx.x & (NC_ - 1);
    stage_tau(tau, b, c, lt);

    const int jd = threadIdx.x;
    if (jd >= JD_) return;
    const int j = jd / D_;
    const int d = jd - j * D_;
    const float* xp = x + ((size_t)b * T_ + (size_t)c * CT_) * JD_ + jd;

    float h = 0.0f, p = 1.0f;
    for (int tt = 0; tt < CT_; ++tt) {
        __builtin_prefetch(xp + (size_t)(tt + 8) * JD_, 0, 1);
        const float tv = lt[(tt >> 2) * J_ + j];
        const float xv = xp[(size_t)tt * JD_];
        h = __fmaf_rn(tv, h - xv, xv);            // (1-tau)*x + tau*h
        p *= tv;
    }
    const int ch = b * J_ + j;
    if (d == 0) P[ch * NC_ + c] = p;
    S[(ch * D_ + d) * NC_ + c] = h;
}

// ---------------------------------------------------------------------------
// Kernel 4 (Phase B): sequential combine across the 16 chunks of a channel.
// h_in[chunk 0] = initial_state; h_in[c+1] = P_c*h_in[c] + S_c.
// ---------------------------------------------------------------------------
__global__ __launch_bounds__(256) void k_combine(const float* __restrict__ init,
                                                 const float* __restrict__ P,
                                                 const float* __restrict__ S,
                                                 float* __restrict__ H) {
    const int idx = blockIdx.x * blockDim.x + threadIdx.x; // (b*J+j)*D+d
    if (idx >= B_ * J_ * D_) return;
    const int chj = idx / D_;                              // b*J + j
    float h = init[idx];                                   // (B,J,D) contiguous
#pragma unroll
    for (int c = 0; c < NC_; ++c) {
        H[idx * NC_ + c] = h;
        h = __fmaf_rn(P[chj * NC_ + c], h, S[idx * NC_ + c]);
    }
}

// ---------------------------------------------------------------------------
// Kernel 5 (Phase C): final scan with correct incoming state; NT stores.
// ---------------------------------------------------------------------------
__global__ __launch_bounds__(352) void k_chunkC(const float* __restrict__ x,
                                                const float* __restrict__ tau,
                                                const float* __restrict__ H,
                                                float* __restrict__ out) {
    __shared__ float lt[TAUW];
    const int b = blockIdx.x >> 4;
    const int c = blockIdx.x & (NC_ - 1);
    stage_tau(tau, b, c, lt);

    const int jd = threadIdx.x;
    if (jd >= JD_) return;
    const int j = jd / D_;
    const int d = jd - j * D_;
    const size_t base = ((size_t)b * T_ + (size_t)c * CT_) * JD_ + jd;
    const float* xp = x + base;
    float* op = out + base;

    float h = H[((b * J_ + j) * D_ + d) * NC_ + c];
    for (int tt = 0; tt < CT_; ++tt) {
        __builtin_prefetch(xp + (size_t)(tt + 8) * JD_, 0, 1);
        const float tv = lt[(tt >> 2) * J_ + j];
        const float xv = xp[(size_t)tt * JD_];
        h = __fmaf_rn(tv, h - xv, xv);
        __builtin_nontemporal_store(h, op + (size_t)tt * JD_);
    }
}

// ---------------------------------------------------------------------------
extern "C" void kernel_launch(void* const* d_in, const int* in_sizes, int n_in,
                              void* d_out, int out_size, void* d_ws, size_t ws_size,
                              hipStream_t stream) {
    const float* poses  = (const float*)d_in[0];   // (B,T,J,D)
    const float* gate   = (const float*)d_in[1];   // (B,Tp)
    const float* logvar = (const float*)d_in[2];   // (B,Tp,Z)
    const float* init   = (const float*)d_in[3];   // (B,J,D)
    float* out = (float*)d_out;

    float* ws    = (float*)d_ws;                   // ~8.8 MB used
    float* sigma = ws + OFF_SIGMA;
    float* tauB  = ws + OFF_TAU;
    float* P     = ws + OFF_P;
    float* S     = ws + OFF_S;
    float* H     = ws + OFF_H;

    // 1) sigma_sq via WMMA row-sum: 32768 rows / 16 rows-per-wave = 2048 waves
    k_sigma<<<256, 256, 0, stream>>>(logvar, sigma);

    // 2) tau at Tp resolution
    const int ntau = B_ * TP_ * J_;
    k_tau<<<(ntau + 255) / 256, 256, 0, stream>>>(gate, sigma, tauB);

    // 3) chunk summaries
    k_chunkA<<<B_ * NC_, 352, 0, stream>>>(poses, tauB, P, S);

    // 4) cross-chunk combine (tiny)
    k_combine<<<(B_ * J_ * D_ + 255) / 256, 256, 0, stream>>>(init, P, S, H);

    // 5) final scan + output
    k_chunkC<<<B_ * NC_, 352, 0, stream>>>(poses, tauB, H, out);
}
